// Recurrent_49486613184597
// MI455X (gfx1250) — matrix operand order repaired and checked by reading, not compiled, not run on hardware
//
#include <hip/hip_runtime.h>

// ---------------- problem constants ----------------
#define LN   2
#define Hh   1024
#define Bb   32
#define Tt   1024
#define G4H  4096      // 4*H
#define K2   2048      // H (input) + H (hidden)
#define NBLK 256       // 2 layers * 2 m0 * 64 jg
#define NWAVE 4        // K-split

typedef __attribute__((ext_vector_type(16))) _Float16 v16h;
typedef __attribute__((ext_vector_type(8)))  _Float16 v8h;
typedef __attribute__((ext_vector_type(8)))  float    v8f;

__device__ __forceinline__ v8f wmma_f16(v16h a, v16h b, v8f c) {
  return __builtin_amdgcn_wmma_f32_16x16x32_f16(false, a, false, b, (short)0, c,
                                                false, false);
}

__device__ __forceinline__ float sigmoidf_(float x) {
  return __builtin_amdgcn_rcpf(1.0f + __expf(-x));
}
__device__ __forceinline__ float tanhf_(float x) {
  return 2.0f * __builtin_amdgcn_rcpf(1.0f + __expf(-2.0f * x)) - 1.0f;
}

// ---------------- pre-pass kernels ----------------
// Wcat[l][n][0:1024] = W_ih[l][n][:], Wcat[l][n][1024:2048] = W_hh[l][n][:]  (f16)
__global__ void k_conv_w(const float* __restrict__ Wih,
                         const float* __restrict__ Whh,
                         _Float16* __restrict__ Wcat) {
  size_t i = (size_t)blockIdx.x * 256 + threadIdx.x;
  if (i >= (size_t)LN * G4H * K2) return;
  int k = (int)(i & (K2 - 1));
  size_t nl = i >> 11;
  int n = (int)(nl & (G4H - 1));
  int l = (int)(nl >> 12);
  float v = (k < Hh) ? Wih[((size_t)l * G4H + n) * Hh + k]
                     : Whh[((size_t)l * G4H + n) * Hh + (k - Hh)];
  Wcat[i] = (_Float16)v;
}

__global__ void k_conv_x(const float* __restrict__ X, _Float16* __restrict__ X16) {
  size_t i = (size_t)blockIdx.x * 256 + threadIdx.x;
  if (i < (size_t)Bb * Tt * Hh) X16[i] = (_Float16)X[i];
}

__global__ void k_bias(const float* __restrict__ bih, const float* __restrict__ bhh,
                       float* __restrict__ bias) {
  int i = blockIdx.x * 256 + threadIdx.x;
  if (i < LN * G4H) bias[i] = bih[i] + bhh[i];
}

__global__ void k_zero(unsigned* __restrict__ p, int n) {
  int i = blockIdx.x * 256 + threadIdx.x;
  if (i < n) p[i] = 0u;
}

// ---------------- persistent LSTM kernel ----------------
// Block = one unit (layer, m0, jg): 16 batch rows x 16 h-columns, all 4 gates.
// 4 waves split K=2048 into 512-chunks; LDS reduce; wave 0 runs the cell.
__global__ __launch_bounds__(128, 1) void lstm_persistent(
    const _Float16* __restrict__ X16,   // [B][T][H] f16
    const _Float16* __restrict__ Wcat,  // [L][4H][2048] f16
    const float* __restrict__ bias,     // [L][4H] (b_ih+b_hh)
    _Float16* __restrict__ h0,          // [2][B][H] f16 double buffer
    _Float16* __restrict__ h1,          // [2][B][H] f16 double buffer
    float* __restrict__ cstate,         // [L][B][H] f32
    unsigned* __restrict__ cnt,         // grid barrier counter
    float* __restrict__ out)            // outputs[B][T][H] ++ hT[L][B][H] ++ cT[L][B][H]
{
  const int bid   = blockIdx.x;
  const int layer = bid >> 7;
  const int m0    = (bid >> 6) & 1;
  const int jg    = bid & 63;
  const int wave  = threadIdx.x >> 5;
  const int lane  = threadIdx.x & 31;
  const int mrow  = lane & 15;   // A row / B column / D column within tile
  const int khalf = lane >> 4;   // which half of the lane group

  __shared__ float red[NWAVE - 1][4][8][32];

  // Per-lane B (weight) row pointers for the 4 gate tiles; includes the
  // lane-dependent k base (khalf*8). Along K: two 8-runs at +0 and +16.
  const _Float16* Wl = Wcat + (size_t)layer * G4H * K2;
  const _Float16* wrow[4];
#pragma unroll
  for (int g = 0; g < 4; ++g)
    wrow[g] = Wl + ((size_t)(g * Hh + jg * 16 + mrow)) * K2 + khalf * 8;

  const int ks0 = wave * 16;          // 16 k-steps of 32 per wave
  const int col = jg * 16 + mrow;     // h column handled by this lane (wave 0)
  const float* biasL = bias + layer * G4H;

  for (int r = 0; r <= Tt; ++r) {
    const bool active = (layer == 0) ? (r < Tt) : (r >= 1);
    const int t  = (layer == 0) ? r : (r - 1);
    const int rb = (r + 1) & 1;       // read buffer (state from round r-1)
    const int wb = r & 1;             // write buffer

    if (active) {
      // A source for this wave's K chunk: waves 0,1 -> input part (k<1024),
      // waves 2,3 -> hidden part (k>=1024). Per-lane row pointer + khalf*8.
      const _Float16* Abase;
      if (wave < 2) {
        if (layer == 0)
          Abase = X16 + ((size_t)(m0 * 16 + mrow)) * Tt * Hh + (size_t)t * Hh;
        else
          Abase = h0 + (size_t)rb * Bb * Hh + (size_t)(m0 * 16 + mrow) * Hh;
      } else {
        const _Float16* hb = (layer == 0) ? h0 : h1;
        Abase = hb + (size_t)rb * Bb * Hh + (size_t)(m0 * 16 + mrow) * Hh;
      }
      const _Float16* ab = Abase + khalf * 8;

      v8f acc[4] = {v8f{0.f}, v8f{0.f}, v8f{0.f}, v8f{0.f}};
#pragma unroll 2
      for (int i = 0; i < 16; ++i) {
        const int ks = ks0 + i;
        const int kl = (ks & 31) << 5;     // local k offset within A source
        v8h alo = *(const v8h*)(ab + kl);
        v8h ahi = *(const v8h*)(ab + kl + 16);
        v16h av;
#pragma unroll
        for (int e = 0; e < 8; ++e) { av[e] = alo[e]; av[e + 8] = ahi[e]; }
        const int kw = ks << 5;            // k offset within concatenated weights
#pragma unroll
        for (int g = 0; g < 4; ++g) {
          v8h blo = *(const v8h*)(wrow[g] + kw);
          v8h bhi = *(const v8h*)(wrow[g] + kw + 16);
          v16h bv;
#pragma unroll
          for (int e = 0; e < 8; ++e) { bv[e] = blo[e]; bv[e + 8] = bhi[e]; }
          acc[g] = wmma_f16(av, bv, acc[g]);
        }
      }

      // K-split reduction across the 4 waves of this block.
      if (wave > 0) {
#pragma unroll
        for (int g = 0; g < 4; ++g)
#pragma unroll
          for (int j = 0; j < 8; ++j) red[wave - 1][g][j][lane] = acc[g][j];
      }
      __syncthreads();

      if (wave == 0) {
#pragma unroll
        for (int g = 0; g < 4; ++g)
#pragma unroll
          for (int j = 0; j < 8; ++j)
            acc[g][j] += red[0][g][j][lane] + red[1][g][j][lane] + red[2][g][j][lane];

        // LSTM cell, fully in registers. D layout: lane -> col, j -> row.
        const float bi = biasL[col];
        const float bf = biasL[Hh + col];
        const float bg = biasL[2 * Hh + col];
        const float bo = biasL[3 * Hh + col];
        float* crow = cstate + (size_t)layer * Bb * Hh;
        _Float16* hw = ((layer == 0) ? h0 : h1) + (size_t)wb * Bb * Hh;
#pragma unroll
        for (int j = 0; j < 8; ++j) {
          const int b = m0 * 16 + khalf * 8 + j;
          const size_t idx = (size_t)b * Hh + col;
          const float iv = sigmoidf_(acc[0][j] + bi);
          const float fv = sigmoidf_(acc[1][j] + bf);
          const float gv = tanhf_(acc[2][j] + bg);
          const float ov = sigmoidf_(acc[3][j] + bo);
          const float cp = crow[idx];
          const float cn = fv * cp + iv * gv;
          const float hv = ov * tanhf_(cn);
          crow[idx] = cn;
          hw[idx] = (_Float16)hv;
          if (layer == 1)
            out[(size_t)b * Tt * Hh + (size_t)t * Hh + col] = hv;
          if (t == Tt - 1) {
            out[(size_t)Bb * Tt * Hh + (size_t)layer * Bb * Hh + idx] = hv;
            out[(size_t)Bb * Tt * Hh + (size_t)LN * Bb * Hh +
                (size_t)layer * Bb * Hh + idx] = cn;
          }
        }
      }
      __syncthreads();   // protect LDS for next round
    }

    // ---- grid-wide barrier (release / spin / acquire) ----
    __builtin_amdgcn_fence(__ATOMIC_RELEASE, "agent");
    __syncthreads();
    if (threadIdx.x == 0) {
      __hip_atomic_fetch_add(cnt, 1u, __ATOMIC_RELEASE, __HIP_MEMORY_SCOPE_AGENT);
      const unsigned target = (unsigned)NBLK * (unsigned)(r + 1);
      while (__hip_atomic_load(cnt, __ATOMIC_ACQUIRE, __HIP_MEMORY_SCOPE_AGENT) <
             target)
        __builtin_amdgcn_s_sleep(1);
    }
    __syncthreads();
    __builtin_amdgcn_fence(__ATOMIC_ACQUIRE, "agent");
  }
}

// ---------------- host launcher ----------------
extern "C" void kernel_launch(void* const* d_in, const int* in_sizes, int n_in,
                              void* d_out, int out_size, void* d_ws, size_t ws_size,
                              hipStream_t stream) {
  const float* X   = (const float*)d_in[0];  // [B,T,H]
  const float* Wih = (const float*)d_in[1];  // [L,4H,H]
  const float* Whh = (const float*)d_in[2];  // [L,4H,H]
  const float* bih = (const float*)d_in[3];  // [L,4H]
  const float* bhh = (const float*)d_in[4];  // [L,4H]
  float* out = (float*)d_out;

  char* ws = (char*)d_ws;
  const size_t wcat_b = (size_t)LN * G4H * K2 * 2;      // 33,554,432
  const size_t x16_b  = (size_t)Bb * Tt * Hh * 2;       // 67,108,864
  const size_t bias_b = (size_t)LN * G4H * 4;           // 32,768
  _Float16* Wcat = (_Float16*)ws;
  _Float16* X16  = (_Float16*)(ws + wcat_b);
  float*    bias = (float*)(ws + wcat_b + x16_b);
  char* zbase    = ws + wcat_b + x16_b + bias_b;        // zeroed every launch
  _Float16* h0   = (_Float16*)zbase;                    // 2*B*H f16 = 131072 B
  _Float16* h1   = (_Float16*)(zbase + 131072);         // 131072 B
  float* cstate  = (float*)(zbase + 262144);            // L*B*H f32 = 262144 B
  unsigned* cnt  = (unsigned*)(zbase + 262144 + 262144);

  const int zero_dwords = (131072 + 131072 + 262144 + 128) / 4;  // 131,104

  k_conv_w<<<(int)((wcat_b / 2 + 255) / 256), 256, 0, stream>>>(Wih, Whh, Wcat);
  k_conv_x<<<(int)((x16_b / 2 + 255) / 256), 256, 0, stream>>>(X, X16);
  k_bias<<<(LN * G4H + 255) / 256, 256, 0, stream>>>(bih, bhh, bias);
  k_zero<<<(zero_dwords + 255) / 256, 256, 0, stream>>>((unsigned*)zbase,
                                                        zero_dwords);
  lstm_persistent<<<NBLK, NWAVE * 32, 0, stream>>>(X16, Wcat, bias, h0, h1,
                                                   cstate, cnt, out);
}
